// ModifiedDecoder_24180665876868
// MI455X (gfx1250) — compile-verified, matrix operand order checked by reading, not run on hardware
//
#include <hip/hip_runtime.h>
#include <hip/hip_bf16.h>
#include <stdint.h>

typedef unsigned short u16;
typedef __attribute__((ext_vector_type(16))) __bf16 v16bf;
typedef __attribute__((ext_vector_type(8)))  float  v8f;
typedef __attribute__((ext_vector_type(4)))  int    v4i;

namespace {
constexpr int kL = 12, kC = 768, kH = 12, kI = 128, kF = 768;
constexpr int kV = 50257, kB = 2, kT = 1024, kHD = 64;
constexpr int kM  = kB * kT;   // 2048 token rows
constexpr int kC3 = 3 * kC;    // 2304
constexpr int kC4 = 4 * kC;    // 3072
}

#define AS1 __attribute__((address_space(1)))
#define AS3 __attribute__((address_space(3)))

#if defined(__has_builtin)
#if __has_builtin(__builtin_amdgcn_global_load_async_to_lds_b128)
#define ASYNC_LDS 1
#endif
#endif

// 16-byte global -> LDS stage: async (ASYNCcnt-tracked, no VGPR round trip)
// via the gfx1250 builtin (signature: (v4i AS1*, v4i AS3*, imm off, imm cpol)),
// else plain vector copy.
__device__ __forceinline__ void stage16(const u16* __restrict__ g,
                                        u16* __restrict__ l) {
#ifdef ASYNC_LDS
  __builtin_amdgcn_global_load_async_to_lds_b128((AS1 v4i*)g, (AS3 v4i*)l, 0,
                                                 0);
#else
  *(uint4*)l = *(const uint4*)g;
#endif
}

__device__ __forceinline__ void async_wait0() {
#ifdef ASYNC_LDS
#if __has_builtin(__builtin_amdgcn_s_wait_asynccnt)
  __builtin_amdgcn_s_wait_asynccnt(0);
#else
  asm volatile("s_wait_asynccnt 0x0" ::: "memory");
#endif
#endif
}

__device__ __forceinline__ u16 f2bf(float f) {
  unsigned u = __builtin_bit_cast(unsigned, f);
  u += 0x7fffu + ((u >> 16) & 1u);          // round-to-nearest-even
  return (u16)(u >> 16);
}

// ---------------------------------------------------------------- converts
__global__ void f32_to_bf16(const float* __restrict__ in, u16* __restrict__ out,
                            size_t n) {
  size_t i = (size_t)blockIdx.x * blockDim.x + threadIdx.x;
  size_t stride = (size_t)gridDim.x * blockDim.x;
  for (; i < n; i += stride) out[i] = f2bf(in[i]);
}

// img[b,c] = sum_f image_features[b,f] * img_w[c,f]
__global__ void img_proj(const float* __restrict__ imgf,
                         const float* __restrict__ img_w,
                         float* __restrict__ imgc) {
  int idx = blockIdx.x * blockDim.x + threadIdx.x;
  if (idx >= kB * kC) return;
  int b = idx / kC, c = idx % kC;
  float s = 0.f;
  for (int f = 0; f < kF; ++f) s += imgf[b * kF + f] * img_w[c * kF + f];
  imgc[idx] = s;
}

// x[b,t,:] = wte[tok] + wpe[t] + img[b]
__global__ __launch_bounds__(256) void embed_kernel(
    const int* __restrict__ tok, const float* __restrict__ wte,
    const float* __restrict__ wpe, const float* __restrict__ imgc,
    float* __restrict__ x) {
  const int row = blockIdx.x;           // b*T + t
  const int b = row / kT, t = row % kT;
  const int id = tok[row];
  for (int c = threadIdx.x; c < kC; c += 256)
    x[(size_t)row * kC + c] =
        wte[(size_t)id * kC + c] + wpe[(size_t)t * kC + c] + imgc[b * kC + c];
}

// ---------------------------------------------------------------- layernorm
__global__ __launch_bounds__(256) void ln_kernel(const float* __restrict__ x,
                                                 const float* __restrict__ g,
                                                 const float* __restrict__ b,
                                                 u16* __restrict__ out) {
  __shared__ float red[256];
  const int r = blockIdx.x, tid = threadIdx.x;
  const float* xr = x + (size_t)r * kC;
  float s = 0.f;
  for (int c = tid; c < kC; c += 256) s += xr[c];
  red[tid] = s; __syncthreads();
  for (int w = 128; w > 0; w >>= 1) {
    if (tid < w) red[tid] += red[tid + w];
    __syncthreads();
  }
  const float mean = red[0] / kC;
  __syncthreads();
  float v = 0.f;
  for (int c = tid; c < kC; c += 256) { float d = xr[c] - mean; v += d * d; }
  red[tid] = v; __syncthreads();
  for (int w = 128; w > 0; w >>= 1) {
    if (tid < w) red[tid] += red[tid + w];
    __syncthreads();
  }
  const float rstd = rsqrtf(red[0] / kC + 1e-5f);
  for (int c = tid; c < kC; c += 256)
    out[(size_t)r * kC + c] = f2bf((xr[c] - mean) * rstd * g[c] + b[c]);
}

// ---------------------------------------------------------------- WMMA GEMM
// C[M,N] = epilogue( A_bf16[M,K] @ B_bf16 (+bias) (+residual_f32) )
// bT==0: B is [K,N] row-major, N must be a multiple of 64 (true for all such
//        calls here); bT==1: B is [N,K] row-major (lm_head / wte^T), ragged N.
// act: 0=none 1=relu 2=gelu(tanh).  Double-buffered LDS, one barrier/K-step.
#define BLK 64
#define BLK_K 32
#define KPAD 40   // halves; 80 B row stride, 16B aligned fragments

__global__ __launch_bounds__(256) void gemm_bf16(
    const u16* __restrict__ A, int lda, const u16* __restrict__ B, int ldb,
    int bT, const float* __restrict__ bias, const float* __restrict__ residual,
    float* __restrict__ outF, u16* __restrict__ outB, int M, int N, int K,
    int act) {
  __shared__ __attribute__((aligned(16))) u16 As[2][BLK][KPAD];   // [m][k]
  __shared__ __attribute__((aligned(16))) u16 Bs[2][BLK][KPAD];   // [n][k]

  const int tid = threadIdx.x;
  const int wave = tid >> 5, lane = tid & 31;
  const int wm = wave & 3, wn = wave >> 2;      // 4x2 wave grid
  const int m0 = blockIdx.y * BLK, n0 = blockIdx.x * BLK;
  const int ml = lane & 15, hi = lane >> 4, hk = hi * 8;

  // staging coordinates
  const int ar = tid >> 2, acg = (tid & 3) * 8;   // 64 rows x 4 k-groups of 8
  const int bkr = tid >> 3, bng = (tid & 7) * 8;  // 32 k-rows x 8 n-groups

  auto stage = [&](int buf, int k0) {
    stage16(A + (size_t)(m0 + ar) * lda + k0 + acg, &As[buf][ar][acg]);
    if (k0 + BLK_K < K)
      __builtin_prefetch(A + (size_t)(m0 + ar) * lda + k0 + BLK_K + acg, 0, 1);
    if (bT) {  // B[n][k]: contiguous in k
      if (n0 + ar < N)
        stage16(B + (size_t)(n0 + ar) * ldb + k0 + acg, &Bs[buf][ar][acg]);
      else
        *(uint4*)&Bs[buf][ar][acg] = make_uint4(0u, 0u, 0u, 0u);
    } else {   // B[k][n]: vector load, transpose-scatter into Bs[n][k]
      union { uint4 u; u16 s[8]; } d;
      d.u = *(const uint4*)(B + (size_t)(k0 + bkr) * ldb + n0 + bng);
#pragma unroll
      for (int j = 0; j < 8; ++j) Bs[buf][bng + j][bkr] = d.s[j];
    }
  };

  v8f accs[2];
#pragma unroll
  for (int t = 0; t < 2; ++t)
#pragma unroll
    for (int j = 0; j < 8; ++j) accs[t][j] = 0.f;

  union Frag { v16bf v; uint4 u[2]; };

  stage(0, 0);
  async_wait0();
  __syncthreads();

  int buf = 0;
  for (int k0 = 0; k0 < K; k0 += BLK_K) {
    if (k0 + BLK_K < K) stage(buf ^ 1, k0 + BLK_K);   // fill next buffer

    Frag fa, fb0, fb1;
    fa.u[0]  = *(const uint4*)&As[buf][wm * 16 + ml][hk];
    fa.u[1]  = *(const uint4*)&As[buf][wm * 16 + ml][hk + 16];
    fb0.u[0] = *(const uint4*)&Bs[buf][wn * 32 + ml][hk];
    fb0.u[1] = *(const uint4*)&Bs[buf][wn * 32 + ml][hk + 16];
    fb1.u[0] = *(const uint4*)&Bs[buf][wn * 32 + 16 + ml][hk];
    fb1.u[1] = *(const uint4*)&Bs[buf][wn * 32 + 16 + ml][hk + 16];
    accs[0] = __builtin_amdgcn_wmma_f32_16x16x32_bf16(
        false, fa.v, false, fb0.v, (short)0, accs[0], false, false);
    accs[1] = __builtin_amdgcn_wmma_f32_16x16x32_bf16(
        false, fa.v, false, fb1.v, (short)0, accs[1], false, false);

    async_wait0();
    __syncthreads();
    buf ^= 1;
  }

  // epilogue: C layout -> lane holds n = ml (+tile), rows m = j + 8*hi
#pragma unroll
  for (int t = 0; t < 2; ++t) {
    const int n = n0 + wn * 32 + t * 16 + ml;
    if (n >= N) continue;
    const float bv = bias ? bias[n] : 0.f;
#pragma unroll
    for (int j = 0; j < 8; ++j) {
      const int m = m0 + wm * 16 + j + 8 * hi;
      float v = accs[t][j] + bv;
      if (act == 1) {
        v = fmaxf(v, 0.f);
      } else if (act == 2) {
        const float c = v + 0.044715f * v * v * v;
        v = 0.5f * v * (1.f + tanhf(0.7978845608f * c));
      }
      if (residual) v += residual[(size_t)m * N + n];
      if (outF) outF[(size_t)m * N + n] = v;
      if (outB) outB[(size_t)m * N + n] = f2bf(v);
    }
  }
}

// ---------------------------------------------------------------- attention
// Flash-style causal attention. One block per (q-tile of 64, head, batch),
// 4 waves; each wave owns 16 q rows. HEAD_DIM = 64.
#define AT_PAD 72   // halves; 144 B row stride, 16B aligned

__global__ __launch_bounds__(128) void attn_kernel(const u16* __restrict__ qkv,
                                                   u16* __restrict__ o) {
  __shared__ __attribute__((aligned(16))) u16 Qs[64][AT_PAD];   // [q][d]
  __shared__ __attribute__((aligned(16))) u16 Ks[64][AT_PAD];   // [kv][d]
  __shared__ __attribute__((aligned(16))) u16 Vs[64][AT_PAD];   // [d][kv]
  __shared__ __attribute__((aligned(16))) u16 Ps[4][16][AT_PAD];// per-wave P

  const int qt = blockIdx.x, h = blockIdx.y, b = blockIdx.z;
  const int tid = threadIdx.x, wave = tid >> 5, lane = tid & 31;
  const int ml = lane & 15, hi = lane >> 4, hk = hi * 8;

  { // stage Q tile (completion covered by async_wait0 before first compute)
    const int r = tid >> 1, cg = (tid & 1) * 32;
    const u16* src = qkv + ((size_t)(b * kT + qt * 64 + r)) * kC3 + h * kHD + cg;
    stage16(src +  0, &Qs[r][cg +  0]);
    stage16(src +  8, &Qs[r][cg +  8]);
    stage16(src + 16, &Qs[r][cg + 16]);
    stage16(src + 24, &Qs[r][cg + 24]);
  }

  float m_i[8], l_i[8];
  v8f accO[4];
#pragma unroll
  for (int j = 0; j < 8; ++j) { m_i[j] = -1e30f; l_i[j] = 0.f; }
#pragma unroll
  for (int t = 0; t < 4; ++t)
#pragma unroll
    for (int j = 0; j < 8; ++j) accO[t][j] = 0.f;

  union Frag { v16bf v; uint4 u[2]; };

  for (int jb = 0; jb <= qt; ++jb) {
    __syncthreads();
    { // stage K (row-major) and V (transposed into [d][kv])
      const int r = tid >> 1, cg = (tid & 1) * 32;
      const size_t rowb = (size_t)(b * kT + jb * 64 + r) * kC3;
      const u16* ks = qkv + rowb + kC + h * kHD + cg;
      stage16(ks +  0, &Ks[r][cg +  0]);
      stage16(ks +  8, &Ks[r][cg +  8]);
      stage16(ks + 16, &Ks[r][cg + 16]);
      stage16(ks + 24, &Ks[r][cg + 24]);
      const u16* vs = qkv + rowb + 2 * kC + h * kHD + cg;
      union { uint4 u; u16 s[8]; } vv;
#pragma unroll
      for (int g2 = 0; g2 < 4; ++g2) {
        vv.u = *(const uint4*)(vs + g2 * 8);
#pragma unroll
        for (int i = 0; i < 8; ++i) Vs[cg + g2 * 8 + i][r] = vv.s[i];
      }
    }
    async_wait0();
    __syncthreads();

    // S = Q @ K^T
    v8f accS[4];
#pragma unroll
    for (int t = 0; t < 4; ++t)
#pragma unroll
      for (int j = 0; j < 8; ++j) accS[t][j] = 0.f;
#pragma unroll
    for (int kk = 0; kk < 64; kk += 32) {
      Frag fq;
      fq.u[0] = *(const uint4*)&Qs[wave * 16 + ml][kk + hk];
      fq.u[1] = *(const uint4*)&Qs[wave * 16 + ml][kk + hk + 16];
#pragma unroll
      for (int t = 0; t < 4; ++t) {
        Frag fk;
        fk.u[0] = *(const uint4*)&Ks[t * 16 + ml][kk + hk];
        fk.u[1] = *(const uint4*)&Ks[t * 16 + ml][kk + hk + 16];
        accS[t] = __builtin_amdgcn_wmma_f32_16x16x32_bf16(
            false, fq.v, false, fk.v, (short)0, accS[t], false, false);
      }
    }

    // scale + causal mask (diagonal tile only)
#pragma unroll
    for (int t = 0; t < 4; ++t)
#pragma unroll
      for (int j = 0; j < 8; ++j) {
        float s = accS[t][j] * 0.125f;   // 1/sqrt(64)
        if (jb == qt) {
          const int ng = t * 16 + ml;
          const int mg = wave * 16 + j + 8 * hi;
          if (ng > mg) s = -1e30f;
        }
        accS[t][j] = s;
      }

    // online softmax: rows span 16 lanes (n = ml) within a half-wave group
#pragma unroll
    for (int j = 0; j < 8; ++j) {
      float mx = fmaxf(fmaxf(accS[0][j], accS[1][j]),
                       fmaxf(accS[2][j], accS[3][j]));
#pragma unroll
      for (int dd = 1; dd < 16; dd <<= 1) mx = fmaxf(mx, __shfl_xor(mx, dd, 16));
      const float mnew = fmaxf(m_i[j], mx);
      const float alpha = __expf(m_i[j] - mnew);
      m_i[j] = mnew;
      float sum = 0.f;
#pragma unroll
      for (int t = 0; t < 4; ++t) {
        const float p = __expf(accS[t][j] - mnew);
        accS[t][j] = p;
        sum += p;
      }
#pragma unroll
      for (int dd = 1; dd < 16; dd <<= 1) sum += __shfl_xor(sum, dd, 16);
      l_i[j] = l_i[j] * alpha + sum;
#pragma unroll
      for (int t = 0; t < 4; ++t) accO[t][j] *= alpha;
    }

    // P -> per-wave LDS region in A-fragment layout
#pragma unroll
    for (int t = 0; t < 4; ++t)
#pragma unroll
      for (int j = 0; j < 8; ++j)
        Ps[wave][j + 8 * hi][t * 16 + ml] = f2bf(accS[t][j]);

    // O += P @ V
#pragma unroll
    for (int kk = 0; kk < 64; kk += 32) {
      Frag fp;
      fp.u[0] = *(const uint4*)&Ps[wave][ml][kk + hk];
      fp.u[1] = *(const uint4*)&Ps[wave][ml][kk + hk + 16];
#pragma unroll
      for (int t2 = 0; t2 < 4; ++t2) {
        Frag fv;
        fv.u[0] = *(const uint4*)&Vs[t2 * 16 + ml][kk + hk];
        fv.u[1] = *(const uint4*)&Vs[t2 * 16 + ml][kk + hk + 16];
        accO[t2] = __builtin_amdgcn_wmma_f32_16x16x32_bf16(
            false, fp.v, false, fv.v, (short)0, accO[t2], false, false);
      }
    }
  }

  // O /= l_i ; write bf16 to [B,T,C]
#pragma unroll
  for (int t2 = 0; t2 < 4; ++t2)
#pragma unroll
    for (int j = 0; j < 8; ++j) {
      const int mg = qt * 64 + wave * 16 + j + 8 * hi;
      const int d = t2 * 16 + ml;
      o[((size_t)(b * kT + mg)) * kC + h * kHD + d] = f2bf(accO[t2][j] / l_i[j]);
    }
}

// ---------------------------------------------------------------- host
extern "C" void kernel_launch(void* const* d_in, const int* in_sizes, int n_in,
                              void* d_out, int out_size, void* d_ws,
                              size_t ws_size, hipStream_t stream) {
  (void)in_sizes; (void)n_in; (void)out_size; (void)ws_size;

  const int*   text  = (const int*)  d_in[0];
  const float* imgf  = (const float*)d_in[1];
  const float* wte   = (const float*)d_in[2];
  const float* wpe   = (const float*)d_in[3];
  const float* img_w = (const float*)d_in[4];
  const float* lnf_g = (const float*)d_in[5];
  const float* lnf_b = (const float*)d_in[6];
  const float* ln1_g = (const float*)d_in[7];
  const float* ln1_b = (const float*)d_in[8];
  const float* ln2_g = (const float*)d_in[9];
  const float* ln2_b = (const float*)d_in[10];
  const float* attn_w = (const float*)d_in[11];
  const float* attn_b = (const float*)d_in[12];
  const float* proj_w = (const float*)d_in[13];
  const float* proj_b = (const float*)d_in[14];
  const float* a1_dw = (const float*)d_in[15];
  const float* a1_db = (const float*)d_in[16];
  const float* a1_uw = (const float*)d_in[17];
  const float* a1_ub = (const float*)d_in[18];
  const float* fc_w  = (const float*)d_in[19];
  const float* fc_b  = (const float*)d_in[20];
  const float* fcp_w = (const float*)d_in[21];
  const float* fcp_b = (const float*)d_in[22];
  const float* a2_dw = (const float*)d_in[23];
  const float* a2_db = (const float*)d_in[24];
  const float* a2_uw = (const float*)d_in[25];
  const float* a2_ub = (const float*)d_in[26];

  char* ws = (char*)d_ws;
  size_t off = 0;
  auto take = [&](size_t bytes) -> char* {
    char* p = ws + off;
    off += (bytes + 255) & ~(size_t)255;
    return p;
  };

  // bf16 weight mirrors
  u16* wte_b   = (u16*)take((size_t)kV * kC * 2);
  u16* attnw_b = (u16*)take((size_t)kL * kC * kC3 * 2);
  u16* projw_b = (u16*)take((size_t)kL * kC * kC * 2);
  u16* fcw_b   = (u16*)take((size_t)kL * kC * kC4 * 2);
  u16* fcpw_b  = (u16*)take((size_t)kL * kC4 * kC * 2);
  u16* a1dw_b  = (u16*)take((size_t)kL * kC * kI * 2);
  u16* a1uw_b  = (u16*)take((size_t)kL * kI * kC * 2);
  u16* a2dw_b  = (u16*)take((size_t)kL * kC * kI * 2);
  u16* a2uw_b  = (u16*)take((size_t)kL * kI * kC * 2);
  // activations
  float* imgc = (float*)take((size_t)kB * kC * 4);
  float* x    = (float*)take((size_t)kM * kC * 4);    // fp32 residual stream
  u16* x_b    = (u16*)take((size_t)kM * kC * 2);      // bf16 mirror of x
  u16* h_b    = (u16*)take((size_t)kM * kC * 2);      // LN outputs
  u16* qkv_b  = (u16*)take((size_t)kM * kC3 * 2);
  u16* att_b  = (u16*)take((size_t)kM * kC * 2);
  u16* t_b    = (u16*)take((size_t)kM * kC4 * 2);
  u16* ad_b   = (u16*)take((size_t)kM * kI * 2);

  auto cvt = [&](const float* src, u16* dst, size_t n) {
    int grid = (int)((n + 256 * 8 - 1) / (256 * 8));
    if (grid > 65535) grid = 65535;
    f32_to_bf16<<<grid, 256, 0, stream>>>(src, dst, n);
  };
  cvt(wte,    wte_b,   (size_t)kV * kC);
  cvt(attn_w, attnw_b, (size_t)kL * kC * kC3);
  cvt(proj_w, projw_b, (size_t)kL * kC * kC);
  cvt(fc_w,   fcw_b,   (size_t)kL * kC * kC4);
  cvt(fcp_w,  fcpw_b,  (size_t)kL * kC4 * kC);
  cvt(a1_dw,  a1dw_b,  (size_t)kL * kC * kI);
  cvt(a1_uw,  a1uw_b,  (size_t)kL * kI * kC);
  cvt(a2_dw,  a2dw_b,  (size_t)kL * kC * kI);
  cvt(a2_uw,  a2uw_b,  (size_t)kL * kI * kC);

  img_proj<<<(kB * kC + 255) / 256, 256, 0, stream>>>(imgf, img_w, imgc);
  embed_kernel<<<kM, 256, 0, stream>>>(text, wte, wpe, imgc, x);

  auto gemm = [&](const u16* A, int lda, const u16* Bm, int ldb, int bT,
                  const float* bias, const float* res, float* oF, u16* oB,
                  int M, int N, int K, int act) {
    dim3 grid((N + BLK - 1) / BLK, (M + BLK - 1) / BLK);
    gemm_bf16<<<grid, 256, 0, stream>>>(A, lda, Bm, ldb, bT, bias, res, oF, oB,
                                        M, N, K, act);
  };

  for (int l = 0; l < kL; ++l) {
    // LN1 -> h
    ln_kernel<<<kM, 256, 0, stream>>>(x, ln1_g + l * kC, ln1_b + l * kC, h_b);
    // qkv = h @ attn_w + attn_b
    gemm(h_b, kC, attnw_b + (size_t)l * kC * kC3, kC3, 0, attn_b + l * kC3,
         nullptr, nullptr, qkv_b, kM, kC3, kC, 0);
    // attention
    {
      dim3 g(kT / 64, kH, kB);
      attn_kernel<<<g, 128, 0, stream>>>(qkv_b, att_b);
    }
    // x = x + att @ proj_w + proj_b
    gemm(att_b, kC, projw_b + (size_t)l * kC * kC, kC, 0, proj_b + l * kC, x, x,
         x_b, kM, kC, kC, 0);
    // adapter 1: x = x + relu(x @ a1_dw + a1_db) @ a1_uw + a1_ub
    gemm(x_b, kC, a1dw_b + (size_t)l * kC * kI, kI, 0, a1_db + l * kI, nullptr,
         nullptr, ad_b, kM, kI, kC, 1);
    gemm(ad_b, kI, a1uw_b + (size_t)l * kI * kC, kC, 0, a1_ub + l * kC, x, x,
         x_b, kM, kC, kI, 0);
    // MLP: x = x + gelu(ln2(x) @ fc_w + fc_b) @ fcp_w + fcp_b
    ln_kernel<<<kM, 256, 0, stream>>>(x, ln2_g + l * kC, ln2_b + l * kC, h_b);
    gemm(h_b, kC, fcw_b + (size_t)l * kC * kC4, kC4, 0, fc_b + l * kC4, nullptr,
         nullptr, t_b, kM, kC4, kC, 2);
    gemm(t_b, kC4, fcpw_b + (size_t)l * kC4 * kC, kC, 0, fcp_b + l * kC, x, x,
         x_b, kM, kC, kC4, 0);
    // adapter 2
    gemm(x_b, kC, a2dw_b + (size_t)l * kC * kI, kI, 0, a2_db + l * kI, nullptr,
         nullptr, ad_b, kM, kI, kC, 1);
    gemm(ad_b, kI, a2uw_b + (size_t)l * kI * kC, kC, 0, a2_ub + l * kC, x, x,
         x_b, kM, kC, kI, 0);
  }

  // final LN + tied lm_head: out = ln_f(x) @ wte^T
  ln_kernel<<<kM, 256, 0, stream>>>(x, lnf_g, lnf_b, h_b);
  gemm(h_b, kC, wte_b, kC, 1, nullptr, nullptr, (float*)d_out, nullptr, kM, kV,
       kC, 0);
}
